// TransformerBlock_76776835383456
// MI455X (gfx1250) — matrix operand order repaired and checked by reading, not compiled, not run on hardware
//
#include <hip/hip_runtime.h>
#include <hip/hip_bf16.h>

// ---------------------------------------------------------------------------
// Transformer block for MI455X (gfx1250): bf16 WMMA GEMMs with TDM
// (tensor_load_to_lds) double-buffered B-tiles + LDS-stripe causal attention.
// B=2, T=2048, E=2048, H=16, D=128, FF=8192.
// ---------------------------------------------------------------------------

typedef __attribute__((ext_vector_type(16))) __bf16 v16bf;
typedef __attribute__((ext_vector_type(8)))  __bf16 v8bf;
typedef __attribute__((ext_vector_type(8)))  float  v8f;
typedef __attribute__((ext_vector_type(4)))  unsigned int v4u;
typedef __attribute__((ext_vector_type(8)))  int v8i;
typedef __attribute__((ext_vector_type(4)))  int v4i;

static constexpr int  Bb   = 2;
static constexpr int  Tt   = 2048;
static constexpr int  Ee   = 2048;
static constexpr int  Hh   = 16;
static constexpr int  Dd   = 128;
static constexpr int  FFf  = 8192;
static constexpr int  MROW = Bb * Tt;         // 4096
static constexpr float LN_EPS = 1e-5f;

// float -> bf16 round-to-nearest-even
static __device__ inline __bf16 f2bf(float f) {
    unsigned int u = __builtin_bit_cast(unsigned int, f);
    unsigned int r = (u + 0x7FFFu + ((u >> 16) & 1u)) >> 16;
    return __builtin_bit_cast(__bf16, (unsigned short)r);
}

static __device__ inline v8f wmma_bf16(v16bf a, v16bf b, v8f c) {
    // D = A(16x32 bf16) * B(32x16 bf16) + C(16x16 f32)
    return __builtin_amdgcn_wmma_f32_16x16x32_bf16(
        /*neg_a=*/false, a, /*neg_b=*/false, b,
        /*c_mod=*/(short)0, c, /*reuse_a=*/false, /*reuse_b=*/false);
}

// A-operand frag (16x32 bf16). Contiguous K order: lane m (low half) holds
// K = {0..15}; lane m (high half) holds K = {16..31}.
// rowbase must already point at  A + row*lda + k0 + (lane>=16 ? 16 : 0).
static __device__ inline v16bf load_a_frag(const __bf16* rowbase) {
    v8bf lo = *(const v8bf*)(rowbase);        // K chunk 0 (8 halves)
    v8bf hi = *(const v8bf*)(rowbase + 8);    // K chunk 1 (+8 halves)
    return __builtin_shufflevector(lo, hi, 0,1,2,3,4,5,6,7,8,9,10,11,12,13,14,15);
}

// ---------------------------------------------------------------------------
// TDM: DMA a 64-row x 32-col bf16 tile (rows strided by K elements) from
// global memory into LDS at byte offset lds_off. 2-D descriptor, groups 2/3
// zero. Tracked by TENSORcnt (per-wave, in-order). This toolchain exposes
// the 6-arg builtin: (u32x4, i32x8, i32x4, i32x4, i32x8, i32 cpol).
// ---------------------------------------------------------------------------
static __device__ inline void tdm_load_b_tile(const __bf16* gsrc,
                                              unsigned lds_off, int K) {
    unsigned long long ga = (unsigned long long)(uintptr_t)gsrc;
    unsigned Ku = (unsigned)K;

    v4u g0;
    g0[0] = 1u;                                   // count=1, user descriptor
    g0[1] = lds_off;                              // lds_addr (bytes)
    g0[2] = (unsigned)(ga & 0xFFFFFFFFull);       // global_addr[31:0]
    g0[3] = (unsigned)((ga >> 32) & 0x01FFFFFFull) | (2u << 30); // [56:32]+type=2

    v8i g1;
    g1[0] = (int)(1u << 16);                      // data_size=1 (2 bytes)
    g1[1] = (int)((Ku & 0xFFFFu) << 16);          // tensor_dim0 lo16 @ [63:48]
    g1[2] = (int)((Ku >> 16) | (64u << 16));      // dim0 hi16, tensor_dim1=64
    g1[3] = (int)(32u << 16);                     // dim1 hi=0, tile_dim0=32
    g1[4] = 64;                                   // tile_dim1=64, tile_dim2=0
    g1[5] = (int)Ku;                              // tensor_dim0_stride[31:0]
    g1[6] = 0;                                    // stride hi16, dim1_stride lo
    g1[7] = 0;

    v4i z4 = {0, 0, 0, 0};
    v8i z8 = {0, 0, 0, 0, 0, 0, 0, 0};
    __builtin_amdgcn_tensor_load_to_lds(g0, g1, z4, z4, z8, /*cpol=*/0);
}

static __device__ inline void wait_tensorcnt0() {
    asm volatile("s_wait_tensorcnt 0x0" ::: "memory");
}
static __device__ inline void wait_tensorcnt1() {
    asm volatile("s_wait_tensorcnt 0x1" ::: "memory");
}
static __device__ inline void wait_dscnt0() {
    asm volatile("s_wait_dscnt 0x0" ::: "memory");
}

// ---------------------------------------------------------------------------
// Positional embedding: x = value + interleaved sin/cos; emit f32 + bf16.
// ---------------------------------------------------------------------------
__global__ void __launch_bounds__(256)
tb_posemb(const float* __restrict__ val, float* __restrict__ xF,
          __bf16* __restrict__ xB) {
    int row = blockIdx.x;               // b*T + t
    int t   = row % Tt;
    const float c = 9.210340371976184f / 1024.0f;  // ln(10000)/half
#pragma unroll
    for (int i = 0; i < 8; ++i) {
        int e  = threadIdx.x + i * 256;
        int hi = e >> 1;
        float invf = expf(-(float)hi * c);
        float ang  = (float)t * invf;
        float emb  = (e & 1) ? cosf(ang) : sinf(ang);
        float v = val[(size_t)row * Ee + e] + emb;
        xF[(size_t)row * Ee + e] = v;
        xB[(size_t)row * Ee + e] = f2bf(v);
    }
}

// ---------------------------------------------------------------------------
// Weight convert + transpose: W[K,N] f32 -> WT[N,K] bf16 (K contiguous).
// ---------------------------------------------------------------------------
__global__ void __launch_bounds__(256)
tb_transpose_cvt(const float* __restrict__ W, __bf16* __restrict__ WT,
                 int K, int N) {
    __shared__ float tile[32][33];
    int nb = blockIdx.x * 32;
    int kb = blockIdx.y * 32;
#pragma unroll
    for (int j = threadIdx.y; j < 32; j += 8)
        tile[j][threadIdx.x] = W[(size_t)(kb + j) * N + (nb + threadIdx.x)];
    __syncthreads();
#pragma unroll
    for (int j = threadIdx.y; j < 32; j += 8)
        WT[(size_t)(nb + j) * K + (kb + threadIdx.x)] =
            f2bf(tile[threadIdx.x][j]);
}

// ---------------------------------------------------------------------------
// Generic bf16 WMMA GEMM: C[M,N] = A[M,K] @ BT[N,K]^T, one wave = 32x64 tile.
// B-tiles (64x32 bf16, 4 KB) are DMA'd into LDS by the Tensor Data Mover,
// double-buffered against the WMMA pipeline (TENSORcnt is per-wave ->
// no barriers needed with a 1-wave workgroup).
// MODE 0: f32 out (+bias)(+residual)            -> Cf32 row-major
// MODE 1: SwiGLU  y=acc+bias; silu(y)*y         -> Cbf16 row-major
// MODE 2: head layout bf16 [B,H,T,D]            -> Cbf16 (Q, K)
// MODE 3: head layout transposed bf16 [B,H,D,T] -> Cbf16 (V)
// ---------------------------------------------------------------------------
template <int MODE>
__global__ void __launch_bounds__(32)
tb_gemm(const __bf16* __restrict__ A, const __bf16* __restrict__ BT,
        const float* __restrict__ bias, const float* __restrict__ resid,
        float* __restrict__ Cf32, __bf16* __restrict__ Cbf16,
        int M, int N, int K) {
    __shared__ __align__(64) __bf16 bTile[2][64 * 32];   // 8 KB double buffer

    const int lane = threadIdx.x & 31;
    const int g    = lane >> 4;       // half-wave selector
    const int n16  = lane & 15;

    const int r0 = blockIdx.y * 32;   // rows
    const int c0 = blockIdx.x * 64;   // cols

    v8f acc[2][4];
#pragma unroll
    for (int r = 0; r < 2; ++r)
#pragma unroll
        for (int c = 0; c < 4; ++c) acc[r][c] = v8f{};

    const __bf16* a0base = A + (size_t)(r0 + n16) * K      + g * 16;
    const __bf16* a1base = A + (size_t)(r0 + 16 + n16) * K + g * 16;
    const __bf16* btile0 = BT + (size_t)c0 * K;           // tile row base

    // Low 32 bits of the flat address of a __shared__ object are the LDS
    // byte offset (aperture mapping).
    const unsigned lds0 = (unsigned)(uintptr_t)(void*)&bTile[0][0];
    const unsigned lds1 = (unsigned)(uintptr_t)(void*)&bTile[1][0];

    const int nk = K / 32;
    tdm_load_b_tile(btile0, lds0, K);                     // prime buffer 0

    for (int ks = 0; ks < nk; ++ks) {
        const int buf = ks & 1;
        if (ks + 1 < nk) {
            // The buffer we are about to overwrite was last read (ds_load)
            // in the previous step -> drain DS before the DMA can land.
            wait_dscnt0();
            tdm_load_b_tile(btile0 + (size_t)(ks + 1) * 32,
                            buf ? lds0 : lds1, K);
            wait_tensorcnt1();    // oldest DMA (current buffer) complete
        } else {
            wait_tensorcnt0();
        }

        v16bf a0 = load_a_frag(a0base + ks * 32);
        v16bf a1 = load_a_frag(a1base + ks * 32);
        const __bf16* lb = buf ? &bTile[1][0] : &bTile[0][0];
#pragma unroll
        for (int nt = 0; nt < 4; ++nt) {
            // B-operand 32x16: lane column n, K contiguous (half g -> K+16)
            const __bf16* bp = lb + (nt * 16 + n16) * 32 + g * 16;
            v16bf b = *(const v16bf*)bp;
            acc[0][nt] = wmma_bf16(a0, b, acc[0][nt]);
            acc[1][nt] = wmma_bf16(a1, b, acc[1][nt]);
        }
    }

    // Epilogue. C/D layout: VGPR i -> M = i + 8*g, N = lane&15.
#pragma unroll
    for (int rt = 0; rt < 2; ++rt) {
#pragma unroll
        for (int nt = 0; nt < 4; ++nt) {
#pragma unroll
            for (int i = 0; i < 8; ++i) {
                int m   = i + (g << 3);
                int row = r0 + rt * 16 + m;
                int col = c0 + nt * 16 + n16;
                float v = acc[rt][nt][i];
                if (MODE == 0) {
                    if (bias)  v += bias[col];
                    if (resid) v += resid[(size_t)row * N + col];
                    Cf32[(size_t)row * N + col] = v;
                } else if (MODE == 1) {
                    float y = v + bias[col];
                    float s = y / (1.0f + __expf(-y));   // silu(y)
                    Cbf16[(size_t)row * N + col] = f2bf(s * y);
                } else {
                    int b = row / Tt, t = row - b * Tt;
                    int h = col / Dd, d = col - h * Dd;
                    if (MODE == 2)
                        Cbf16[(((size_t)(b * Hh + h) * Tt + t) * Dd) + d] = f2bf(v);
                    else
                        Cbf16[(((size_t)(b * Hh + h) * Dd + d) * Tt) + t] = f2bf(v);
                }
            }
        }
    }
}

// ---------------------------------------------------------------------------
// Causal attention, one wave per (b,h, 16-row tile). The full 16 x T score
// stripe lives in LDS (16*2048*4 = 128 KB; CDNA5 WGP has 320 KB).
//  q,k: bf16 [B,H,T,D]   v: bf16 [B,H,D,T] (transposed)   o: bf16 [B,T,E]
// ---------------------------------------------------------------------------
__global__ void __launch_bounds__(32)
tb_attention(const __bf16* __restrict__ qb, const __bf16* __restrict__ kb,
             const __bf16* __restrict__ vt, __bf16* __restrict__ ob) {
    extern __shared__ float sS[];               // [16][Tt]
    const int lane = threadIdx.x & 31;
    const int g    = lane >> 4;
    const int n16  = lane & 15;

    const int ntiles = Tt / 16;                 // 128
    const int it = blockIdx.x % ntiles;         // row tile
    const int bh = blockIdx.x / ntiles;         // b*H + h

    const __bf16* q_ = qb + (size_t)bh * Tt * Dd;
    const __bf16* k_ = kb + (size_t)bh * Tt * Dd;
    const __bf16* v_ = vt + (size_t)bh * Dd * Tt;

    // Preload Q row-tile frags (D=128 -> 4 K-steps of 32)
    v16bf aQ[4];
#pragma unroll
    for (int ks = 0; ks < 4; ++ks)
        aQ[ks] = load_a_frag(q_ + (size_t)(it * 16 + n16) * Dd + ks * 32 + g * 16);

    const float scale = 0.08838834764831845f;   // 1/sqrt(128)

    // S stripe = Q @ K^T with causal mask, scaled
    for (int jt = 0; jt <= it; ++jt) {
        v8f c = v8f{};
#pragma unroll
        for (int ks = 0; ks < 4; ++ks) {
            const __bf16* bp = k_ + (size_t)(jt * 16 + n16) * Dd + ks * 32 + g * 16;
            v16bf b = *(const v16bf*)bp;
            c = wmma_bf16(aQ[ks], b, c);
        }
#pragma unroll
        for (int i = 0; i < 8; ++i) {
            int m    = i + (g << 3);
            int trow = it * 16 + m;
            int scol = jt * 16 + n16;
            float vv = (scol <= trow) ? c[i] * scale : -1e30f;
            sS[(size_t)m * Tt + scol] = vv;
        }
    }

    // Softmax over the stripe: lanes (n16, n16+16) cooperate on row n16.
    const int ncols = (it + 1) * 16;
    {
        int m = n16;
        float mx = -1e30f;
        for (int j = g; j < ncols; j += 2) mx = fmaxf(mx, sS[(size_t)m * Tt + j]);
        mx = fmaxf(mx, __shfl_xor(mx, 16, 32));
        float sum = 0.0f;
        for (int j = g; j < ncols; j += 2) {
            float e = __expf(sS[(size_t)m * Tt + j] - mx);
            sS[(size_t)m * Tt + j] = e;
            sum += e;
        }
        sum += __shfl_xor(sum, 16, 32);
        float inv = 1.0f / sum;
        for (int j = g; j < ncols; j += 2) sS[(size_t)m * Tt + j] *= inv;
    }

    // O = P @ V, K-dim = ncols in steps of 32 (pad last step with zeros)
    v8f oc[8];
#pragma unroll
    for (int nt = 0; nt < 8; ++nt) oc[nt] = v8f{};

    const int nsteps = (it + 2) >> 1;           // ceil(ncols/32)
    for (int step = 0; step < nsteps; ++step) {
        v16bf aP;
#pragma unroll
        for (int j = 0; j < 16; ++j) {
            int kl = j + g * 16;  // contiguous A-frag K mapping
            int kk = step * 32 + kl;
            float pv = (kk < ncols) ? sS[(size_t)n16 * Tt + kk] : 0.0f;
            aP[j] = f2bf(pv);
        }
#pragma unroll
        for (int nt = 0; nt < 8; ++nt) {
            const __bf16* bp = v_ + (size_t)(nt * 16 + n16) * Tt + step * 32 + g * 16;
            v16bf b = *(const v16bf*)bp;
            oc[nt] = wmma_bf16(aP, b, oc[nt]);
        }
    }

    // Store O to [B,T,E] bf16 (A-operand for the Wo GEMM)
    const int b_ = bh / Hh, h_ = bh % Hh;
#pragma unroll
    for (int nt = 0; nt < 8; ++nt) {
#pragma unroll
        for (int i = 0; i < 8; ++i) {
            int m    = i + (g << 3);
            int trow = it * 16 + m;
            int e    = h_ * Dd + nt * 16 + n16;
            ob[((size_t)b_ * Tt + trow) * Ee + e] = f2bf(oc[nt][i]);
        }
    }
}

// ---------------------------------------------------------------------------
// LayerNorm over E=2048 per row; optional bf16 copy of output.
// ---------------------------------------------------------------------------
__global__ void __launch_bounds__(256)
tb_layernorm(const float* __restrict__ in, const float* __restrict__ gain,
             float* __restrict__ outF, __bf16* __restrict__ outB) {
    __shared__ float red[256];
    const int row = blockIdx.x;
    const int tid = threadIdx.x;
    const float* r = in + (size_t)row * Ee;

    float s = 0.0f;
#pragma unroll
    for (int i = 0; i < 8; ++i) s += r[tid + i * 256];
    red[tid] = s;
    __syncthreads();
    for (int o = 128; o > 0; o >>= 1) {
        if (tid < o) red[tid] += red[tid + o];
        __syncthreads();
    }
    float mu = red[0] / (float)Ee;
    __syncthreads();

    float vs = 0.0f;
#pragma unroll
    for (int i = 0; i < 8; ++i) {
        float d = r[tid + i * 256] - mu;
        vs += d * d;
    }
    red[tid] = vs;
    __syncthreads();
    for (int o = 128; o > 0; o >>= 1) {
        if (tid < o) red[tid] += red[tid + o];
        __syncthreads();
    }
    float rstd = rsqrtf(red[0] / (float)Ee + LN_EPS);

#pragma unroll
    for (int i = 0; i < 8; ++i) {
        int e = tid + i * 256;
        float v = (r[e] - mu) * rstd * gain[e];
        outF[(size_t)row * Ee + e] = v;
        if (outB) outB[(size_t)row * Ee + e] = f2bf(v);
    }
}

// ---------------------------------------------------------------------------
// Host-side orchestration
// ---------------------------------------------------------------------------
extern "C" void kernel_launch(void* const* d_in, const int* in_sizes, int n_in,
                              void* d_out, int out_size, void* d_ws, size_t ws_size,
                              hipStream_t stream) {
    (void)in_sizes; (void)n_in; (void)out_size; (void)ws_size;

    const float* value = (const float*)d_in[0];
    // d_in[1] = mask (causal; applied analytically)
    const float* Wq = (const float*)d_in[2];
    const float* Wk = (const float*)d_in[3];
    const float* Wv = (const float*)d_in[4];
    const float* Wo = (const float*)d_in[5];
    const float* bo = (const float*)d_in[6];
    const float* W1 = (const float*)d_in[7];
    const float* b1 = (const float*)d_in[8];
    const float* W2 = (const float*)d_in[9];
    const float* b2 = (const float*)d_in[10];
    const float* g1 = (const float*)d_in[11];
    const float* g2 = (const float*)d_in[12];
    float* out = (float*)d_out;

    // Workspace carve-up (aliased; ~252 MB total)
    char* p = (char*)d_ws;
    auto take = [&](size_t bytes) -> void* {
        void* r = (void*)p;
        p += (bytes + 255) & ~(size_t)255;
        return r;
    };
    float*  xF    = (float*)take((size_t)MROW * Ee * 4);   // x f32, later h f32
    __bf16* xB    = (__bf16*)take((size_t)MROW * Ee * 2);  // x bf16, later h bf16
    __bf16* slotW = (__bf16*)take((size_t)FFf * Ee * 2);   // one WT at a time
    __bf16* qB    = (__bf16*)take((size_t)MROW * Ee * 2);  // [B,H,T,D]
    __bf16* kB    = (__bf16*)take((size_t)MROW * Ee * 2);  // [B,H,T,D]
    __bf16* vT    = (__bf16*)take((size_t)MROW * Ee * 2);  // [B,H,D,T]
    __bf16* oB    = (__bf16*)take((size_t)MROW * Ee * 2);  // [B,T,E]
    float*  sF    = (float*)take((size_t)MROW * Ee * 4);   // pre-LN sums (x2)
    __bf16* actB  = (__bf16*)take((size_t)MROW * FFf * 2); // SwiGLU output

    // 1) x = value + pos_emb  (f32 + bf16)
    tb_posemb<<<MROW, 256, 0, stream>>>(value, xF, xB);

    const dim3 tblk(32, 8);
    const dim3 gblk(32);
    const dim3 gemmE(Ee / 64, MROW / 32);   // N=2048
    const dim3 gemmF(FFf / 64, MROW / 32);  // N=8192

    // 2) Q = x @ Wq -> bf16 [B,H,T,D]
    tb_transpose_cvt<<<dim3(Ee / 32, Ee / 32), tblk, 0, stream>>>(Wq, slotW, Ee, Ee);
    tb_gemm<2><<<gemmE, gblk, 0, stream>>>(xB, slotW, nullptr, nullptr,
                                           nullptr, qB, MROW, Ee, Ee);
    // 3) K
    tb_transpose_cvt<<<dim3(Ee / 32, Ee / 32), tblk, 0, stream>>>(Wk, slotW, Ee, Ee);
    tb_gemm<2><<<gemmE, gblk, 0, stream>>>(xB, slotW, nullptr, nullptr,
                                           nullptr, kB, MROW, Ee, Ee);
    // 4) V -> transposed [B,H,D,T]
    tb_transpose_cvt<<<dim3(Ee / 32, Ee / 32), tblk, 0, stream>>>(Wv, slotW, Ee, Ee);
    tb_gemm<3><<<gemmE, gblk, 0, stream>>>(xB, slotW, nullptr, nullptr,
                                           nullptr, vT, MROW, Ee, Ee);

    // 5) Causal attention (LDS stripe = 16*T*4 = 128 KB per workgroup)
    tb_attention<<<Bb * Hh * (Tt / 16), 32, 16 * Tt * sizeof(float), stream>>>(
        qB, kB, vT, oB);

    // 6) attn_out = o @ Wo + bo + x   (f32 pre-LN sum)
    tb_transpose_cvt<<<dim3(Ee / 32, Ee / 32), tblk, 0, stream>>>(Wo, slotW, Ee, Ee);
    tb_gemm<0><<<gemmE, gblk, 0, stream>>>(oB, slotW, bo, xF,
                                           sF, nullptr, MROW, Ee, Ee);

    // 7) h = LN(sF, g1) -> f32 (reuse xF) + bf16 (reuse xB)
    tb_layernorm<<<MROW, 256, 0, stream>>>(sF, g1, xF, xB);

    // 8) act = silu(h@W1+b1) * (h@W1+b1) -> bf16 [M, FF]
    tb_transpose_cvt<<<dim3(FFf / 32, Ee / 32), tblk, 0, stream>>>(W1, slotW, Ee, FFf);
    tb_gemm<1><<<gemmF, gblk, 0, stream>>>(xB, slotW, b1, nullptr,
                                           nullptr, actB, MROW, FFf, Ee);

    // 9) ff = act @ W2 + b2 + h  (f32 pre-LN sum, reuse sF)
    tb_transpose_cvt<<<dim3(Ee / 32, FFf / 32), tblk, 0, stream>>>(W2, slotW, FFf, Ee);
    tb_gemm<0><<<gemmE, gblk, 0, stream>>>(actB, slotW, b2, xF,
                                           sF, nullptr, MROW, Ee, FFf);

    // 10) out = LN(sF, g2)
    tb_layernorm<<<MROW, 256, 0, stream>>>(sF, g2, out, nullptr);
}
